// DDGCRN_39376260170239
// MI455X (gfx1250) — matrix operand order repaired and checked by reading, not compiled, not run on hardware
//
#include <hip/hip_runtime.h>
#include <hip/hip_bf16.h>
#include <math.h>

typedef __attribute__((ext_vector_type(16))) _Float16 v16h;
typedef __attribute__((ext_vector_type(8)))  float    v8f;
typedef __attribute__((ext_vector_type(4)))  unsigned int v4u;
typedef __attribute__((ext_vector_type(8)))  int      v8i;
typedef __attribute__((ext_vector_type(4)))  int      v4i;

#define DEV __device__ __forceinline__

static constexpr int cB=32, cT=12, cN=307, cNp=320, cD=64, cC=65, cKp=160;
static constexpr int cR  = cB*cN*cT;   // 117888 attention rows
static constexpr int cBN = cB*cN;      // 9824

struct H8 { _Float16 h[8]; };

// ---- CDNA5 WMMA fragment index helpers (16x16x32 f16) ----
DEV int a_k(int i,int lane){ return i + ((lane>>4)<<4); }
DEV int b_k(int i,int lane){ return i + ((lane>>4)<<4); }

DEV v8f wmma16(v16h a, v16h b, v8f c){
  return __builtin_amdgcn_wmma_f32_16x16x32_f16(false,a,false,b,(short)0,c,false,false);
}
DEV float wred32(float v){
  #pragma unroll
  for(int m=16;m>0;m>>=1) v += __shfl_xor(v,m,32);
  return v;
}
DEV float wred8(float v){  // reduce within groups of 8 lanes (one attention head)
  #pragma unroll
  for(int m=4;m>0;m>>=1) v += __shfl_xor(v,m,32);
  return v;
}
DEV float sigm(float x){ return 1.f/(1.f+expf(-x)); }
DEV float pe_val(int t,int d){
  int j=d>>1;
  float dv = expf(-(logf(10000.f)/(float)cD)*(float)(2*j));
  float a = (float)t*dv;
  return (d&1)? cosf(a): sinf(a);
}

// fragment-linear index for B-operand tiles: element (kb, col) of a Kp x o tile
// dst = (((kt*ntiles)+nt)*32 + lane)*16 + i  with kb = kt*32 + i + 16*(lane>>4), col = nt*16+(lane&15)
DEV int fragidx(int kb,int col,int ntiles){
  int kt = kb>>5, kbl = kb&31;
  int lanehi = kbl>>4, i = kbl&15;
  int nt = col>>4, ln = (lanehi<<4)|(col&15);
  return ((kt*ntiles+nt)*32 + ln)*16 + i;
}

// ---------- precompute pooled per-node weights (fragment-linear f16) ----------
__global__ void k_pool(const float* __restrict__ ne_s, const float* __restrict__ wpool,
                       const float* __restrict__ bpool, _Float16* __restrict__ W,
                       float* __restrict__ Bias, int o, int total){
  int id = blockIdx.x*blockDim.x + threadIdx.x;
  if(id>=total) return;
  int oc = id % o; int kk = (id/o) % cKp; int n = id/(o*cKp);
  float v = 0.f;
  if(kk < 2*cC){
    int ks = (kk>=cC)?1:0; int i = kk - cC*ks;
    #pragma unroll
    for(int e=0;e<10;e++) v += ne_s[n*10+e]*wpool[((e*2+ks)*cC+i)*o+oc];
  }
  W[(size_t)n*cKp*o + fragidx(kk,oc,o>>4)] = (_Float16)v;
  if(kk==0){
    float bv=0.f;
    #pragma unroll
    for(int e=0;e<10;e++) bv += ne_s[n*10+e]*bpool[e*o+oc];
    Bias[n*o+oc]=bv;
  }
}

__global__ void k_zero(float* p, int count){
  int id = blockIdx.x*blockDim.x + threadIdx.x;
  if(id<count) p[id]=0.f;
}

// ---------- per-timestep prep: xs=[x,(z*)state], nodevec=tanh(ne1*filt(MLP(xs))) ----------
__global__ void k_prep(const float* __restrict__ src, int t,
                       const float* __restrict__ last, const float* __restrict__ ec2w,
                       const float* __restrict__ ec2b, int use_res,
                       const float* __restrict__ state, const float* __restrict__ gate_pre, int zmode,
                       const float* __restrict__ w1,const float* __restrict__ b1,
                       const float* __restrict__ w2,const float* __restrict__ b2,
                       const float* __restrict__ w3,const float* __restrict__ b3,
                       const float* __restrict__ ne_s,const float* __restrict__ tod_emb,
                       const float* __restrict__ dow_emb,
                       float* __restrict__ xs, _Float16* __restrict__ nv){
  int id = blockIdx.x*blockDim.x + threadIdx.x;
  if(id >= cB*cNp) return;
  int b = id/cNp, pn = id%cNp;
  float* xrow = xs + (size_t)id*80;
  _Float16* nrow = nv + (size_t)id*32;
  if(pn >= cN){
    for(int i=0;i<80;i++) xrow[i]=0.f;
    for(int i=0;i<32;i++) nrow[i]=(_Float16)0.f;
    return;
  }
  int n = pn;
  const float* s3 = src + ((size_t)(b*cT+t)*cN+n)*3;
  float xv = s3[0];
  if(use_res){
    float acc = ec2b[t];
    for(int d=0;d<cD;d++) acc += last[(b*cN+n)*cD+d]*ec2w[t*cD+d];
    xv -= acc;
  }
  xrow[0]=xv;
  for(int d=0;d<cD;d++){
    float sd = state[(b*cN+n)*cD+d];
    if(zmode){
      float z = sigm(gate_pre[(size_t)(b*cN+n)*128+d]);
      sd *= z;
    }
    xrow[1+d]=sd;
  }
  for(int i=cC;i<80;i++) xrow[i]=0.f;
  float h1[16];
  #pragma unroll
  for(int j=0;j<16;j++) h1[j]=b1[j];
  for(int i=0;i<cC;i++){
    float v = xrow[i];
    #pragma unroll
    for(int j=0;j<16;j++) h1[j] += v*w1[i*16+j];
  }
  #pragma unroll
  for(int j=0;j<16;j++) h1[j]=sigm(h1[j]);
  float h2a=b2[0], h2b=b2[1];
  #pragma unroll
  for(int j=0;j<16;j++){ h2a+=h1[j]*w2[j*2]; h2b+=h1[j]*w2[j*2+1]; }
  h2a=sigm(h2a); h2b=sigm(h2b);
  int tod = (int)(s3[1]*288.f); tod = tod<0?0:(tod>287?287:tod);
  int dow = (int)(s3[2]);       dow = dow<0?0:(dow>6?6:dow);
  #pragma unroll
  for(int e=0;e<10;e++){
    float filt = h2a*w3[e] + h2b*w3[10+e] + b3[e];
    float ne = ne_s[n*10+e]*tod_emb[tod*10+e]*dow_emb[dow*10+e];
    nrow[e] = (_Float16)tanhf(ne*filt);
  }
  #pragma unroll
  for(int e=10;e<32;e++) nrow[e]=(_Float16)0.f;
}

// ---------- xs transpose to f16 [b][80][320] for contiguous B-frag loads ----------
__global__ void k_xst(const float* __restrict__ xs, _Float16* __restrict__ xsT){
  int id = blockIdx.x*blockDim.x + threadIdx.x;
  if(id >= cB*cNp*80) return;
  int c = id % 80; int pn = (id/80)%cNp; int b = id/(80*cNp);
  xsT[((size_t)b*80 + c)*cNp + pn] = (_Float16)xs[((size_t)b*cNp+pn)*80 + c];
}

// ---------- adj = relu(nv @ nv^T) + I   (WMMA, K=E padded to 32) ----------
__global__ void k_adj(const _Float16* __restrict__ nv, float* __restrict__ adj){
  int gid = blockIdx.x*8 + (threadIdx.x>>5);
  int lane = threadIdx.x & 31;
  int jt = gid % 20; int it = (gid/20)%20; int b = gid/400;
  if(b >= cB) return;
  const _Float16* nvb = nv + (size_t)b*cNp*32;
  int m = it*16 + (lane&15);
  int n = jt*16 + (lane&15);
  int off8 = (lane>>4)<<3;
  const H8* pa0 = (const H8*)(nvb + m*32 + off8);
  const H8* pa1 = (const H8*)(nvb + m*32 + off8 + 16);
  const H8* pb0 = (const H8*)(nvb + n*32 + off8);
  const H8* pb1 = pb0 + 2;
  v16h a, bb;
  #pragma unroll
  for(int i=0;i<8;i++){
    a[i]   = pa0->h[i];  a[8+i]  = pa1->h[i];
    bb[i]  = pb0->h[i];  bb[8+i] = pb1->h[i];
  }
  v8f c = {};
  c = wmma16(a,bb,c);
  #pragma unroll
  for(int r=0;r<8;r++){
    int row = it*16 + r + ((lane>>4)<<3);
    int col = jt*16 + (lane&15);
    float v = c[r]; v = v>0.f? v:0.f;
    if(row==col) v += 1.f;
    adj[((size_t)b*cNp+row)*cNp + col] = v;
  }
}

// ---------- d = rsqrt(rowsum(adj)) ----------
__global__ void k_rowsum(const float* __restrict__ adj, float* __restrict__ dsum){
  int gid = blockIdx.x*8 + (threadIdx.x>>5);
  int lane = threadIdx.x&31;
  if(gid >= cB*cNp) return;
  const float* r = adj + (size_t)gid*cNp;
  float s=0.f;
  for(int c=lane;c<cNp;c+=32) s += r[c];
  s = wred32(s);
  if(lane==0) dsum[gid] = rsqrtf(s);
}

// ---------- xg2 = (D adj D) @ xs  (WMMA, K=320; vectorized A, contiguous B via xsT) ----------
__global__ void k_xg2(const float* __restrict__ adj, const float* __restrict__ dsum,
                      const _Float16* __restrict__ xsT, float* __restrict__ xg2){
  int gid = blockIdx.x*8 + (threadIdx.x>>5);
  int lane = threadIdx.x&31;
  int nt = gid%5; int mt=(gid/5)%20; int b=gid/100;
  if(b>=cB) return;
  const float* adjb = adj + (size_t)b*cNp*cNp;
  const float* db   = dsum + b*cNp;
  const _Float16* xtb = xsT + (size_t)b*80*cNp;
  int m   = mt*16 + (lane&15);
  int col = nt*16 + (lane&15);
  int off8 = (lane>>4)<<3;
  int hi16 = (lane>>4)<<4;
  v8f acc = {};
  for(int kt=0; kt<10; kt++){
    int kbase = kt*32 + off8;
    if(kt<9) __builtin_prefetch(&adjb[(size_t)m*cNp + kbase + 32], 0, 1);
    const float4* qa0 = (const float4*)(adjb + (size_t)m*cNp + kbase);
    const float4* qd0 = (const float4*)(db + kbase);
    const float4* qa1 = (const float4*)(adjb + (size_t)m*cNp + kbase + 16);
    const float4* qd1 = (const float4*)(db + kbase + 16);
    float4 A0=qa0[0], A1=qa0[1], D0=qd0[0], D1=qd0[1];
    float4 A2=qa1[0], A3=qa1[1], D2=qd1[0], D3=qd1[1];
    v16h a, bb;
    a[0]=(_Float16)(A0.x*D0.x); a[1]=(_Float16)(A0.y*D0.y); a[2]=(_Float16)(A0.z*D0.z); a[3]=(_Float16)(A0.w*D0.w);
    a[4]=(_Float16)(A1.x*D1.x); a[5]=(_Float16)(A1.y*D1.y); a[6]=(_Float16)(A1.z*D1.z); a[7]=(_Float16)(A1.w*D1.w);
    a[8]=(_Float16)(A2.x*D2.x); a[9]=(_Float16)(A2.y*D2.y); a[10]=(_Float16)(A2.z*D2.z); a[11]=(_Float16)(A2.w*D2.w);
    a[12]=(_Float16)(A3.x*D3.x); a[13]=(_Float16)(A3.y*D3.y); a[14]=(_Float16)(A3.z*D3.z); a[15]=(_Float16)(A3.w*D3.w);
    const H8* pb0 = (const H8*)(xtb + (size_t)col*cNp + kt*32 + off8);
    const H8* pb1 = pb0 + 2;
    #pragma unroll
    for(int i=0;i<8;i++){ bb[i]=pb0->h[i]; bb[8+i]=pb1->h[i]; }
    acc = wmma16(a,bb,acc);
  }
  #pragma unroll
  for(int r=0;r<8;r++){
    int row = mt*16 + r + ((lane>>4)<<3);
    xg2[((size_t)b*cNp+row)*80+col] = acc[r]*db[row];
  }
}

// ---------- per-node projection with TDM-staged weights in LDS ----------
__global__ void k_dgcnout(const float* __restrict__ xs, const float* __restrict__ xg2,
                          const _Float16* __restrict__ W, const float* __restrict__ bias,
                          float* __restrict__ outp, int o){
  __shared__ _Float16 sW[cKp*128];   // up to 40KB, fragment-linear
  int node = blockIdx.x;
  int wave = threadIdx.x>>5; int lane = threadIdx.x&31;
  int ntiles = o>>4;
  int nelem = cKp*o;
#if __has_builtin(__builtin_amdgcn_tensor_load_to_lds) && __has_builtin(__builtin_amdgcn_s_wait_tensorcnt)
  if(wave==0){
    unsigned long long ga = (unsigned long long)(const void*)(W + (size_t)node*nelem);
    unsigned elems = (unsigned)nelem;
    v4u g0 = {0,0,0,0};
    g0[0] = 1u;                                            // count=1 (valid user D#)
    g0[1] = 0u;                                            // lds_addr = 0 (sW is LDS base)
    g0[2] = (unsigned)(ga & 0xFFFFFFFFu);                  // global_addr[31:0]
    g0[3] = (unsigned)((ga >> 32) & 0x1FFFFFFu) | (2u<<30);// global_addr[56:32], type=2
    v8i g1 = {0,0,0,0,0,0,0,0};
    g1[0] = (int)(1u<<16);                                 // data_size=1 (2 bytes)
    g1[1] = (int)((elems & 0xFFFFu)<<16);                  // tensor_dim0[15:0]
    g1[2] = (int)(((elems>>16)&0xFFFFu) | (1u<<16));       // tensor_dim0[31:16], tensor_dim1=1
    g1[3] = (int)((elems & 0xFFFFu)<<16);                  // tile_dim0 (1-D tile)
    g1[5] = (int)elems;                                    // tensor_dim0_stride[31:0]
    v4i gz = {0,0,0,0};
#if __clang_major__ >= 23
    v8i gz8 = {0,0,0,0,0,0,0,0};
    __builtin_amdgcn_tensor_load_to_lds(g0,g1,gz,gz,gz8,0);
#else
    __builtin_amdgcn_tensor_load_to_lds(g0,g1,gz,gz,0);
#endif
    __builtin_amdgcn_s_wait_tensorcnt((short)0);
  }
#else
  for(int i=threadIdx.x; i<nelem; i+=256) sW[i] = W[(size_t)node*nelem + i];
#endif
  __syncthreads();
  int njobs = 2*ntiles;
  for(int j=wave; j<njobs; j+=8){
    int mt = j & 1; int nt = j >> 1;
    int bidx = mt*16 + (lane&15);     // batch index = A row
    int col  = nt*16 + (lane&15);
    v8f acc={};
    for(int kt=0;kt<5;kt++){
      v16h a,bb;
      #pragma unroll
      for(int i=0;i<16;i++){
        int ka = kt*32 + a_k(i,lane);
        float av;
        if(ka < cC)        av = xs [((size_t)bidx*cNp+node)*80 + ka];
        else if(ka < 2*cC) av = xg2[((size_t)bidx*cNp+node)*80 + (ka-cC)];
        else               av = 0.f;
        a[i] = (_Float16)av;
      }
      const H8* pb0 = (const H8*)(sW + ((kt*ntiles+nt)*32 + lane)*16);
      const H8* pb1 = pb0 + 1;
      #pragma unroll
      for(int i=0;i<8;i++){ bb[i]=pb0->h[i]; bb[8+i]=pb1->h[i]; }
      acc = wmma16(a,bb,acc);
    }
    #pragma unroll
    for(int r=0;r<8;r++){
      int bat = mt*16 + r + ((lane>>4)<<3);
      outp[((size_t)bat*cN+node)*o + col] = acc[r] + bias[node*o+col];
    }
  }
}

// ---------- GRU state update ----------
__global__ void k_state(const float* __restrict__ gate_pre, const float* __restrict__ upd_pre,
                        float* __restrict__ state, float* __restrict__ extra){
  int id = blockIdx.x*blockDim.x + threadIdx.x;
  if(id >= cB*cN*cD) return;
  int d = id & 63; int bn = id >> 6;
  float r  = sigm(gate_pre[(size_t)bn*128 + 64 + d]);
  float hc = tanhf(upd_pre[id]);
  float s  = state[id];
  float ns = r*s + (1.f-r)*hc;
  state[id] = ns;
  if(extra) extra[id] = ns;
}

// ---------- MHA: build broadcast PE rows ----------
__global__ void k_buildxq(const float* __restrict__ last, _Float16* __restrict__ xq){
  int id = blockIdx.x*blockDim.x + threadIdx.x;
  if(id >= cR*cD) return;
  int d = id & 63; int row = id >> 6;
  int t = row % cT; int bn = row / cT;
  xq[id] = (_Float16)(last[bn*cD+d] + pe_val(t,d));
}

// ---------- generic WMMA GEMM: A[M x 64](f16) @ W[64 x 64] (LDS fragment-linear) ----------
__global__ void k_proj(const _Float16* __restrict__ A, const float* __restrict__ W,
                       const float* __restrict__ bias, int act,
                       const float* __restrict__ resid,
                       float* __restrict__ out32, _Float16* __restrict__ out16, int M){
  __shared__ _Float16 sW[64*64];
  for(int idx=threadIdx.x; idx<4096; idx+=256){
    int kb = idx>>6, col = idx&63;
    sW[fragidx(kb,col,4)] = (_Float16)W[idx];
  }
  __syncthreads();
  int wave = threadIdx.x>>5, lane = threadIdx.x&31;
  int mbase = (blockIdx.x*8+wave)*16;
  if(mbase >= M) return;
  int row = mbase + (lane&15);
  int off8 = (lane>>4)<<3;
  for(int nt=0; nt<4; nt++){
    int col = nt*16 + (lane&15);
    v8f acc={};
    for(int kt=0;kt<2;kt++){
      v16h a,bb;
      if(row < M){
        const H8* pa0 = (const H8*)(A + (size_t)row*64 + kt*32 + 2*off8);
        const H8* pa1 = (const H8*)(A + (size_t)row*64 + kt*32 + 2*off8 + 8);
        #pragma unroll
        for(int i=0;i<8;i++){ a[i]=pa0->h[i]; a[8+i]=pa1->h[i]; }
      } else {
        #pragma unroll
        for(int i=0;i<16;i++) a[i]=(_Float16)0.f;
      }
      const H8* pb0 = (const H8*)(sW + ((kt*4+nt)*32 + lane)*16);
      const H8* pb1 = pb0 + 1;
      #pragma unroll
      for(int i=0;i<8;i++){ bb[i]=pb0->h[i]; bb[8+i]=pb1->h[i]; }
      acc = wmma16(a,bb,acc);
    }
    #pragma unroll
    for(int r=0;r<8;r++){
      int rr = mbase + r + ((lane>>4)<<3);
      if(rr<M){
        float v = acc[r];
        if(bias)  v += bias[col];
        if(act)   v = v>0.f? v:0.f;
        if(resid) v += resid[(size_t)rr*64+col];
        if(out32) out32[(size_t)rr*64+col]=v;
        if(out16) out16[(size_t)rr*64+col]=(_Float16)v;
      }
    }
  }
}

// ---------- attention at query t=11 (4 heads x 16), + residual + LN1 ----------
__global__ void k_attn(const float* __restrict__ last, const float* __restrict__ wq,
                       const _Float16* __restrict__ Kh, const _Float16* __restrict__ Vh,
                       const float* __restrict__ g1, const float* __restrict__ bln1,
                       float* __restrict__ ctx_out, _Float16* __restrict__ ln1h){
  __shared__ float sx[8][64];
  int wave=threadIdx.x>>5, lane=threadIdx.x&31;
  int bn = blockIdx.x*8+wave;   // grid is exact: cBN/8
  int d0 = lane*2;
  float x0 = last[bn*cD+d0]   + pe_val(11,d0);
  float x1 = last[bn*cD+d0+1] + pe_val(11,d0+1);
  sx[wave][d0]=x0; sx[wave][d0+1]=x1;
  __syncthreads();
  float q0=0.f, q1=0.f;
  for(int c=0;c<64;c++){
    float xc = sx[wave][c];
    q0 += xc*wq[c*64+d0];
    q1 += xc*wq[c*64+d0+1];
  }
  const _Float16* Kb = Kh + (size_t)bn*cT*cD;
  const _Float16* Vb = Vh + (size_t)bn*cT*cD;
  float sc[12]; float mx = -1e30f;
  #pragma unroll
  for(int k=0;k<12;k++){
    float p = q0*(float)Kb[k*64+d0] + q1*(float)Kb[k*64+d0+1];
    p = wred8(p) * 0.125f;           // per-head (8-lane) reduce; /sqrt(64)
    sc[k]=p; mx = p>mx? p:mx;
  }
  float ssum=0.f;
  #pragma unroll
  for(int k=0;k<12;k++){ sc[k]=expf(sc[k]-mx); ssum+=sc[k]; }
  float inv = 1.f/ssum;
  float c0=x0, c1=x1;
  #pragma unroll
  for(int k=0;k<12;k++){
    float a = sc[k]*inv;
    c0 += a*(float)Vb[k*64+d0];
    c1 += a*(float)Vb[k*64+d0+1];
  }
  ctx_out[(size_t)bn*64+d0]=c0; ctx_out[(size_t)bn*64+d0+1]=c1;
  float mean = wred32(c0+c1)*(1.f/64.f);
  float dx0=c0-mean, dx1=c1-mean;
  float var = wred32(dx0*dx0+dx1*dx1)*(1.f/64.f);
  float rs = rsqrtf(var+1e-5f);
  ln1h[(size_t)bn*64+d0]   = (_Float16)(dx0*rs*g1[d0]+bln1[d0]);
  ln1h[(size_t)bn*64+d0+1] = (_Float16)(dx1*rs*g1[d0+1]+bln1[d0+1]);
}

// ---------- LN2 + out1 + out2 (end-convs) ----------
__global__ void k_final(const float* __restrict__ o2, const float* __restrict__ g2,
                        const float* __restrict__ b2ln,
                        const float* __restrict__ h2last, const float* __restrict__ last,
                        const float* __restrict__ ec1w, const float* __restrict__ ec1b,
                        const float* __restrict__ ec3w, const float* __restrict__ ec3b,
                        float* __restrict__ out){
  int wave=threadIdx.x>>5, lane=threadIdx.x&31;
  int bn = blockIdx.x*8+wave;
  int b = bn/cN, n = bn%cN;
  int d0 = lane*2;
  float v0=o2[(size_t)bn*64+d0], v1=o2[(size_t)bn*64+d0+1];
  float mean = wred32(v0+v1)*(1.f/64.f);
  float dx0=v0-mean, dx1=v1-mean;
  float var = wred32(dx0*dx0+dx1*dx1)*(1.f/64.f);
  float rs = rsqrtf(var+1e-5f);
  float t0 = dx0*rs*g2[d0]  +b2ln[d0];
  float t1 = dx1*rs*g2[d0+1]+b2ln[d0+1];
  float h0 = h2last[bn*64+d0]  +t0;
  float h1v= h2last[bn*64+d0+1]+t1;
  float l0 = last[bn*64+d0], l1=last[bn*64+d0+1];
  for(int h=0;h<12;h++){
    float p = l0*ec1w[h*64+d0] + l1*ec1w[h*64+d0+1]
            + h0*ec3w[h*64+d0] + h1v*ec3w[h*64+d0+1];
    p = wred32(p);
    if(lane==0) out[(size_t)(b*12+h)*cN+n] = p + ec1b[h] + ec3b[h];
  }
}

// ===================== host =====================
struct DG { const float *w1,*b1,*w2,*b2,*w3,*b3,*wpool,*bpool; };

extern "C" void kernel_launch(void* const* d_in, const int* in_sizes, int n_in,
                              void* d_out, int out_size, void* d_ws, size_t ws_size,
                              hipStream_t stream){
  (void)in_sizes; (void)n_in; (void)out_size; (void)ws_size;
  const float* src     = (const float*)d_in[0];
  const float* ne_s    = (const float*)d_in[1];
  const float* tod_emb = (const float*)d_in[2];
  const float* dow_emb = (const float*)d_in[3];
  auto getdg=[&](int o){ DG g; g.w1=(const float*)d_in[o]; g.b1=(const float*)d_in[o+1];
    g.w2=(const float*)d_in[o+2]; g.b2=(const float*)d_in[o+3]; g.w3=(const float*)d_in[o+4];
    g.b3=(const float*)d_in[o+5]; g.wpool=(const float*)d_in[o+6]; g.bpool=(const float*)d_in[o+7];
    return g; };
  DG g1=getdg(4), u1=getdg(12), g2d=getdg(20), u2=getdg(28);
  const float* wq  =(const float*)d_in[36];
  const float* wk  =(const float*)d_in[37];
  const float* wv  =(const float*)d_in[38];
  const float* ln1g=(const float*)d_in[39];
  const float* ln1b=(const float*)d_in[40];
  const float* ln2g=(const float*)d_in[41];
  const float* ln2b=(const float*)d_in[42];
  const float* f1w =(const float*)d_in[43];
  const float* f1b =(const float*)d_in[44];
  const float* f2w =(const float*)d_in[45];
  const float* f2b =(const float*)d_in[46];
  const float* ec1w=(const float*)d_in[47];
  const float* ec1b=(const float*)d_in[48];
  const float* ec2w=(const float*)d_in[49];
  const float* ec2b=(const float*)d_in[50];
  const float* ec3w=(const float*)d_in[51];
  const float* ec3b=(const float*)d_in[52];

  char* base = (char*)d_ws;
  size_t off = 0;
  auto alloc=[&](size_t n){ size_t o=off; off += (n+255)&~(size_t)255; return o; };
  // persistent
  _Float16* Wg1=(_Float16*)(base+alloc((size_t)cN*cKp*128*2));
  _Float16* Wu1=(_Float16*)(base+alloc((size_t)cN*cKp*64*2));
  _Float16* Wg2=(_Float16*)(base+alloc((size_t)cN*cKp*128*2));
  _Float16* Wu2=(_Float16*)(base+alloc((size_t)cN*cKp*64*2));
  float* Bg1=(float*)(base+alloc((size_t)cN*128*4));
  float* Bu1=(float*)(base+alloc((size_t)cN*64*4));
  float* Bg2=(float*)(base+alloc((size_t)cN*128*4));
  float* Bu2=(float*)(base+alloc((size_t)cN*64*4));
  float* state=(float*)(base+alloc((size_t)cB*cN*cD*4));
  float* last =(float*)(base+alloc((size_t)cB*cN*cD*4));
  size_t sb = off;
  // encoder scratch (aliased with MHA scratch; disjoint in time)
  size_t cur = sb;
  auto nxt=[&](size_t n){ size_t o=cur; cur += (n+255)&~(size_t)255; return o; };
  float*    xs      =(float*)   (base+nxt((size_t)cB*cNp*80*4));
  _Float16* nv      =(_Float16*)(base+nxt((size_t)cB*cNp*32*2));
  float*    adj     =(float*)   (base+nxt((size_t)cB*cNp*cNp*4));
  float*    dsum    =(float*)   (base+nxt((size_t)cB*cNp*4));
  float*    xg2     =(float*)   (base+nxt((size_t)cB*cNp*80*4));
  _Float16* xsT     =(_Float16*)(base+nxt((size_t)cB*80*cNp*2));
  float*    gate_pre=(float*)   (base+nxt((size_t)cB*cN*128*4));
  float*    upd_pre =(float*)   (base+nxt((size_t)cB*cN*64*4));
  // MHA scratch (same region)
  cur = sb;
  _Float16* xq_h =(_Float16*)(base+nxt((size_t)cR*cD*2));
  _Float16* Kh   =(_Float16*)(base+nxt((size_t)cR*cD*2));
  _Float16* Vh   =(_Float16*)(base+nxt((size_t)cR*cD*2));
  float*    ctx  =(float*)   (base+nxt((size_t)cBN*cD*4));
  _Float16* ln1h =(_Float16*)(base+nxt((size_t)cBN*cD*2));
  _Float16* ffn1h=(_Float16*)(base+nxt((size_t)cBN*cD*2));
  float*    o2b  =(float*)   (base+nxt((size_t)cBN*cD*4));

  // 1) precompute pooled per-node weights (time/batch invariant; fragment-linear)
  {
    int tg = cN*cKp*128, tu = cN*cKp*64;
    k_pool<<<(tg+255)/256,256,0,stream>>>(ne_s,g1.wpool ,g1.bpool ,Wg1,Bg1,128,tg);
    k_pool<<<(tu+255)/256,256,0,stream>>>(ne_s,u1.wpool ,u1.bpool ,Wu1,Bu1, 64,tu);
    k_pool<<<(tg+255)/256,256,0,stream>>>(ne_s,g2d.wpool,g2d.bpool,Wg2,Bg2,128,tg);
    k_pool<<<(tu+255)/256,256,0,stream>>>(ne_s,u2.wpool ,u2.bpool ,Wu2,Bu2, 64,tu);
  }

  auto run_dgcn=[&](const DG& P,int t,int use_res,int zmode,
                    const _Float16* W,const float* Bias,int o,float* outpre){
    k_prep<<<(cB*cNp+255)/256,256,0,stream>>>(src,t,last,ec2w,ec2b,use_res,
        state,gate_pre,zmode,P.w1,P.b1,P.w2,P.b2,P.w3,P.b3,
        ne_s,tod_emb,dow_emb,xs,nv);
    k_xst   <<<(cB*cNp*80+255)/256,256,0,stream>>>(xs,xsT);
    k_adj   <<<(cB*400+7)/8,256,0,stream>>>(nv,adj);
    k_rowsum<<<(cB*cNp+7)/8,256,0,stream>>>(adj,dsum);
    k_xg2   <<<(cB*100+7)/8,256,0,stream>>>(adj,dsum,xsT,xg2);
    k_dgcnout<<<cN,256,0,stream>>>(xs,xg2,W,Bias,outpre,o);
  };
  auto run_enc=[&](const DG& G,const DG& U,const _Float16* Wg,const float* Bg,
                   const _Float16* Wu,const float* Bu,int use_res,float* extra){
    int nst = cB*cN*cD;
    k_zero<<<(nst+255)/256,256,0,stream>>>(state,nst);
    for(int t=0;t<cT;t++){
      run_dgcn(G,t,use_res,0,Wg,Bg,128,gate_pre);
      run_dgcn(U,t,use_res,1,Wu,Bu, 64,upd_pre);
      float* ex = (t==cT-1)? extra : nullptr;
      k_state<<<(nst+255)/256,256,0,stream>>>(gate_pre,upd_pre,state,ex);
    }
  };

  // 2) encoder 1 -> last ; 3) encoder 2 (x - endconv(last,ec2)) -> state == h2_last
  run_enc(g1 ,u1, Wg1,Bg1, Wu1,Bu1, 0, last);
  run_enc(g2d,u2, Wg2,Bg2, Wu2,Bu2, 1, nullptr);

  // 4) MHA over PE-broadcast of `last`; only query position t=11 survives
  k_buildxq<<<(cR*cD)/256,256,0,stream>>>(last,xq_h);
  k_proj<<<(cR+127)/128,256,0,stream>>>(xq_h,wk,nullptr,0,nullptr,nullptr,Kh,cR);
  k_proj<<<(cR+127)/128,256,0,stream>>>(xq_h,wv,nullptr,0,nullptr,nullptr,Vh,cR);
  k_attn<<<cBN/8,256,0,stream>>>(last,wq,Kh,Vh,ln1g,ln1b,ctx,ln1h);
  k_proj<<<(cBN+127)/128,256,0,stream>>>(ln1h ,f1w,f1b,1,nullptr,nullptr,ffn1h,cBN);
  k_proj<<<(cBN+127)/128,256,0,stream>>>(ffn1h,f2w,f2b,0,ctx,o2b,nullptr,cBN);

  // 5) LN2 + out1 + out2
  k_final<<<cBN/8,256,0,stream>>>(o2b,ln2g,ln2b,state,last,ec1w,ec1b,ec3w,ec3b,(float*)d_out);
}
